// ImprovedHomogeneousMoELayer_82145544503967
// MI455X (gfx1250) — compile-verified, matrix operand order checked by reading, not run on hardware
//
#include <hip/hip_runtime.h>

// ---------------------------------------------------------------------------
// MI455X (gfx1250) implementation. wave32, WMMA 16x16x32 bf16 -> f32.
// Compute-bound (~260 GFLOP vs ~230MB unique bytes => ~10us HBM floor).
// All large GEMMs on v_wmma_f32_16x16x32_bf16; weights pre-packed into
// fragment order so every WMMA operand load is a 128-bit load. GEMM wave
// tile is 32x128: two A fragments share each B fragment (16 WMMA / 20 b128).
// ---------------------------------------------------------------------------

typedef unsigned short bhalf;
typedef __attribute__((ext_vector_type(16))) __bf16 v16bf;
typedef __attribute__((ext_vector_type(8)))  __bf16 v8bf;
typedef __attribute__((ext_vector_type(8)))  float  v8f;

constexpr int CB = 8;      // batch
constexpr int CS = 1024;   // seq
constexpr int CD = 512;    // model dim
constexpr int CDFF = 2048; // ffn dim
constexpr int CE = 8;      // experts
constexpr int CK = 2;      // top-k
constexpr int CH = 4;      // heads
constexpr int CDH = CD / CH;          // 128
constexpr long long CBS  = (long long)CB * CS;       // 8192
constexpr long long CBSD = CBS * CD;                 // 4194304
constexpr int CPAIRS = CB * CK;                      // 16

// ------------------------------ helpers ------------------------------------

__device__ __forceinline__ bhalf f2bfu(float f) {
  union { float f; unsigned u; } c; c.f = f;
  unsigned r = c.u + 0x7FFFu + ((c.u >> 16) & 1u);   // round-to-nearest-even
  return (bhalf)(r >> 16);
}

__device__ __forceinline__ float geluf(float x) {
  return 0.5f * x * (1.0f + erff(x * 0.70710678118654752f));
}

__device__ __forceinline__ v8f wmma_bf16(v16bf a, v16bf b, v8f c) {
  return __builtin_amdgcn_wmma_f32_16x16x32_bf16(
      /*neg_a=*/false, a, /*neg_b=*/false, b,
      /*c_mod=*/(short)0, c, /*reuse_a=*/false, /*reuse_b=*/false);
}

// Build a 16-element bf16 fragment from two 16-byte chunks (2x b128 loads).
// 16-bit fragment layout: elements 0..7 = K[8*half .. 8*half+7],
// elements 8..15 = K[16+8*half .. 16+8*half+7] (within a 32-wide K chunk).
__device__ __forceinline__ v16bf ld_frag2(const bhalf* p0, const bhalf* p1) {
  v8bf lo = *(const v8bf*)p0;
  v8bf hi = *(const v8bf*)p1;
  return __builtin_shufflevector(lo, hi, 0, 1, 2, 3, 4, 5, 6, 7,
                                 8, 9, 10, 11, 12, 13, 14, 15);
}

// ------------------------------ f32 -> bf16 --------------------------------

__global__ void cvt_kernel(const float* __restrict__ in,
                           bhalf* __restrict__ out, long long n) {
  long long i = (long long)blockIdx.x * blockDim.x + threadIdx.x;
  long long stride = (long long)gridDim.x * blockDim.x;
  for (; i < n; i += stride) out[i] = f2bfu(in[i]);
}

// ------------------------- weight fragment packing -------------------------
// Repack f32 weight [Kd,N] (row-major, per expert) into bf16 fragment order:
// dst[e][kc][nt][lane][el] contiguous, so a B fragment is 32 bytes/lane and
// a full fragment-block is one coalesced 1KB stream per wave.

__global__ void pack_w_kernel(const float* __restrict__ src,
                              bhalf* __restrict__ dst,
                              int Kd, int N, long long total) {
  const long long per = (long long)Kd * N;
  const int NT = N >> 4;
  long long i = (long long)blockIdx.x * blockDim.x + threadIdx.x;
  const long long stride = (long long)gridDim.x * blockDim.x;
  for (; i < total; i += stride) {
    const long long e = i / per;
    const long long r = i % per;
    const int el = (int)(r & 15);
    const int lane = (int)((r >> 4) & 31);
    const long long rest = r >> 9;
    const int nt = (int)(rest % NT);
    const int kc = (int)(rest / NT);
    const int j = el >> 1, s = el & 1;
    const int half = lane >> 4, lm = lane & 15;
    const int kk = kc * 32 + ((j & 4) << 2) + (half << 3) + ((j & 3) << 1) + s;
    const int n = nt * 16 + lm;
    dst[i] = f2bfu(src[e * per + (long long)kk * N + n]);
  }
}

// ------------------------------ WMMA GEMM ----------------------------------
// One wave computes a 32x128 output tile: two A fragments (rows m0..m0+15 and
// m0+16..m0+31) share each of the 8 B fragments -> 16 WMMAs per 20 b128
// loads per K-chunk. Weights come from pack_w_kernel fragment order.

__global__ __launch_bounds__(32) void gemm_bf16_kernel(
    const bhalf* __restrict__ A, const bhalf* __restrict__ W,
    const float* __restrict__ bias, const float* __restrict__ residual,
    const int* __restrict__ eidx,
    float* __restrict__ C, bhalf* __restrict__ Cbf,
    int M, int N, int Kd,
    long long aBatchStride, int aDiv,
    long long wStride, long long biasStride,
    long long cBatchStride, int act) {
  const int lane = threadIdx.x;
  const int half = lane >> 4, lm = lane & 15;
  const int n0 = blockIdx.x * 128;
  const int m0 = blockIdx.y * 32;
  const int z  = blockIdx.z;
  const int NT = N >> 4;

  const long long e = eidx ? (long long)eidx[z] : (long long)z;
  const bhalf* Ap = A + (long long)(z / aDiv) * aBatchStride;
  const bhalf* Wp = W + e * wStride;
  const float* bp = bias ? bias + e * biasStride : nullptr;
  float* Cp  = C   ? C   + (long long)z * cBatchStride : nullptr;
  bhalf* Cbp = Cbf ? Cbf + (long long)z * cBatchStride : nullptr;
  const float* rp = residual ? residual + (long long)z * cBatchStride : nullptr;

  v8f acc0[8] = {};
  v8f acc1[8] = {};
  const bhalf* arow0 = Ap + (long long)(m0 + lm) * Kd;
  const bhalf* arow1 = Ap + (long long)(m0 + 16 + lm) * Kd;

  for (int k0 = 0; k0 < Kd; k0 += 32) {
    const bhalf* pb = Wp + ((long long)(k0 >> 5) * NT + (n0 >> 4)) * 512
                      + (lane << 4);
    // speculative prefetch of next K-chunk streams (unconditional: OOB
    // speculative prefetches are dropped by hardware)
    __builtin_prefetch(arow0 + k0 + 32, 0, 1);
    __builtin_prefetch(arow1 + k0 + 32, 0, 1);
    __builtin_prefetch(pb + (long long)NT * 512, 0, 1);
    const v16bf a0 = ld_frag2(arow0 + k0 + (half << 3),
                              arow0 + k0 + 16 + (half << 3));
    const v16bf a1 = ld_frag2(arow1 + k0 + (half << 3),
                              arow1 + k0 + 16 + (half << 3));
#pragma unroll
    for (int t = 0; t < 8; ++t) {
      const v16bf bw = ld_frag2(pb + t * 512, pb + t * 512 + 8);
      acc0[t] = wmma_bf16(a0, bw, acc0[t]);
      acc1[t] = wmma_bf16(a1, bw, acc1[t]);
    }
  }

#pragma unroll
  for (int t = 0; t < 8; ++t) {
    const int n = n0 + t * 16 + lm;
    const float bv = bp ? bp[n] : 0.0f;
#pragma unroll
    for (int v = 0; v < 8; ++v) {
#pragma unroll
      for (int g = 0; g < 2; ++g) {
        const int m = m0 + g * 16 + v + (half << 3);
        const long long off = (long long)m * N + n;
        float val = (g ? acc1[t][v] : acc0[t][v]) + bv;
        if (rp) val += rp[off];
        if (act == 1)      val = geluf(val);
        else if (act == 2) val = 1.0f / (1.0f + expf(-val));
        if (Cp)  Cp[off]  = val;
        if (Cbp) Cbp[off] = f2bfu(val);
      }
    }
  }
}

// ------------------------- V transpose for attention -----------------------
// vt[b][h][dh][s] = qkv[b*S+s][2D + h*DH + dh] so P@V B-fragments are
// contiguous b128 loads.

__global__ void vtrans_kernel(const bhalf* __restrict__ qkv,
                              bhalf* __restrict__ vt) {
  const long long total = (long long)CB * CH * CDH * CS;
  long long i = (long long)blockIdx.x * blockDim.x + threadIdx.x;
  const long long stride = (long long)gridDim.x * blockDim.x;
  for (; i < total; i += stride) {
    const int s = (int)(i % CS);
    long long r = i / CS;
    const int d = (int)(r % CDH);
    r /= CDH;
    const int h = (int)(r % CH);
    const int b = (int)(r / CH);
    vt[i] = qkv[((long long)b * CS + s) * (3 * CD) + 2 * CD + h * CDH + d];
  }
}

// ------------------------------ flash attention ----------------------------
// grid (S/16, H, B), block = 1 wave. Q tile resident in 4 A-fragments,
// online softmax over 32-key chunks via LDS, QK^T and PV on WMMA. All
// fragment feeds are b128 (global) or ds_load_b128 (P from LDS).

__global__ __launch_bounds__(32) void attn_kernel(
    const bhalf* __restrict__ qkv, const bhalf* __restrict__ vt,
    bhalf* __restrict__ oattn) {
  const int lane = threadIdx.x;
  const int half = lane >> 4, lm = lane & 15;
  const int q0 = blockIdx.x * 16;
  const int h  = blockIdx.y;
  const int b  = blockIdx.z;
  const long long rs = 3 * CD;                   // qkv row stride
  const bhalf* base = qkv + (long long)b * CS * rs;
  const int qcol = h * CDH;
  const int kcol = CD + h * CDH;
  const bhalf* vtb = vt + ((long long)b * CH + h) * CDH * CS;

  __shared__ float sS[16 * 32];
  __shared__ __align__(16) bhalf sP[16 * 32];
  __shared__ float sM[16], sL[16], sScale[16];

  if (lane < 16) { sM[lane] = -1e30f; sL[lane] = 0.0f; }
  __syncthreads();

  // resident Q fragments (dh = 128 = 4 chunks of K=32)
  const bhalf* qrow = base + (long long)(q0 + lm) * rs + qcol;
  v16bf aq[4];
#pragma unroll
  for (int c = 0; c < 4; ++c)
    aq[c] = ld_frag2(qrow + c * 32 + (half << 3),
                     qrow + c * 32 + 16 + (half << 3));

  v8f acc[8] = {};                                // O tile 16 x 128
  const float sc = 0.0883883476483184f;           // 1/sqrt(128)

  for (int kb = 0; kb < CS; kb += 32) {
    // --- scores: two 16x16 tiles over 32 keys ---
    const bhalf* krow0 = base + (long long)(kb + lm) * rs + kcol;
    const bhalf* krow1 = base + (long long)(kb + 16 + lm) * rs + kcol;
    v8f s0 = {}, s1 = {};
#pragma unroll
    for (int c = 0; c < 4; ++c) {
      const v16bf b0 = ld_frag2(krow0 + c * 32 + (half << 3),
                                krow0 + c * 32 + 16 + (half << 3));
      const v16bf b1 = ld_frag2(krow1 + c * 32 + (half << 3),
                                krow1 + c * 32 + 16 + (half << 3));
      s0 = wmma_bf16(aq[c], b0, s0);
      s1 = wmma_bf16(aq[c], b1, s1);
    }
#pragma unroll
    for (int v = 0; v < 8; ++v) {
      const int r = v + (half << 3);
      sS[r * 32 + lm]      = s0[v] * sc;
      sS[r * 32 + 16 + lm] = s1[v] * sc;
    }
    __syncthreads();

    // --- online softmax per row (threads 0..15) ---
    if (lane < 16) {
      float m = sM[lane];
      float cm = m;
      for (int i = 0; i < 32; ++i) cm = fmaxf(cm, sS[lane * 32 + i]);
      const float scl = expf(m - cm);
      float sum = 0.0f;
      for (int i = 0; i < 32; ++i) {
        float p = expf(sS[lane * 32 + i] - cm);
        sum += p;
        sP[lane * 32 + i] = f2bfu(p);
      }
      sL[lane] = sL[lane] * scl + sum;
      sM[lane] = cm;
      sScale[lane] = scl;
    }
    __syncthreads();

    // --- rescale accumulators ---
    float f[8];
#pragma unroll
    for (int v = 0; v < 8; ++v) f[v] = sScale[v + (half << 3)];
#pragma unroll
    for (int t = 0; t < 8; ++t)
#pragma unroll
      for (int v = 0; v < 8; ++v) acc[t][v] = acc[t][v] * f[v];

    // --- P fragment from LDS (2x ds_load_b128) ---
    const v16bf ap = ld_frag2(sP + lm * 32 + (half << 3),
                              sP + lm * 32 + 16 + (half << 3));

    // --- O += P @ V (B fragments from transposed V, b128 loads) ---
#pragma unroll
    for (int t = 0; t < 8; ++t) {
      const bhalf* vrow = vtb + (long long)(t * 16 + lm) * CS + kb;
      const v16bf bv = ld_frag2(vrow + (half << 3), vrow + 16 + (half << 3));
      acc[t] = wmma_bf16(ap, bv, acc[t]);
    }
    __syncthreads();
  }

#pragma unroll
  for (int v = 0; v < 8; ++v) {
    const int r = v + (half << 3);
    const float inv = 1.0f / sL[r];
    const long long row = (long long)b * CS + q0 + r;
#pragma unroll
    for (int t = 0; t < 8; ++t)
      oattn[row * CD + h * CDH + t * 16 + lm] = f2bfu(acc[t][v] * inv);
  }
}

// ------------------------------ LayerNorm ----------------------------------

__global__ __launch_bounds__(256) void ln_kernel(
    const float* __restrict__ X, const float* __restrict__ residual,
    const float* __restrict__ gamma, const float* __restrict__ beta,
    const int* __restrict__ eidx, int rowsPerPair, long long gStride,
    int N, int postGelu, float* __restrict__ outF, bhalf* __restrict__ outB) {
  const int row = blockIdx.x;
  const int tid = threadIdx.x;
  const long long e = eidx ? (long long)eidx[row / rowsPerPair] : 0;
  const float* gp = gamma + e * gStride;
  const float* bp = beta + e * gStride;
  const float* xr = X + (long long)row * N;
  const float* rr = residual ? residual + (long long)row * N : nullptr;

  __shared__ float red[256];
  __shared__ float sMean, sRstd;

  float s = 0.0f, ss = 0.0f;
  for (int c = tid; c < N; c += 256) {
    float v = xr[c] + (rr ? rr[c] : 0.0f);
    s += v; ss += v * v;
  }
  red[tid] = s; __syncthreads();
  for (int o = 128; o > 0; o >>= 1) {
    if (tid < o) red[tid] += red[tid + o];
    __syncthreads();
  }
  if (tid == 0) sMean = red[0] / N;
  __syncthreads();
  red[tid] = ss; __syncthreads();
  for (int o = 128; o > 0; o >>= 1) {
    if (tid < o) red[tid] += red[tid + o];
    __syncthreads();
  }
  if (tid == 0) {
    const float m = sMean;
    sRstd = rsqrtf(red[0] / N - m * m + 1e-5f);
  }
  __syncthreads();
  const float m = sMean, r = sRstd;
  for (int c = tid; c < N; c += 256) {
    float v = xr[c] + (rr ? rr[c] : 0.0f);
    v = (v - m) * r * gp[c] + bp[c];
    if (postGelu) v = geluf(v);
    const long long off = (long long)row * N + c;
    if (outF) outF[off] = v;
    if (outB) outB[off] = f2bfu(v);
  }
}

// ------------------------------ pooled mean --------------------------------

__global__ void pooled_kernel(const float* __restrict__ ctx,
                              float* __restrict__ pooled) {
  const int i = blockIdx.x * blockDim.x + threadIdx.x;   // < B*D
  const int b = i / CD, d = i % CD;
  const float* p = ctx + (long long)b * CS * CD + d;
  float s = 0.0f;
  for (int t = 0; t < CS; ++t) s += p[(long long)t * CD];
  pooled[i] = s * (1.0f / CS);
}

// ------------------------------ gate MLP -----------------------------------

__global__ __launch_bounds__(256) void gate_mlp_kernel(
    const float* __restrict__ pooled,
    const float* __restrict__ g1w, const float* __restrict__ g1b,
    const float* __restrict__ lng, const float* __restrict__ lnb,
    const float* __restrict__ g2w, const float* __restrict__ g2b,
    const float* __restrict__ g3w, const float* __restrict__ g3b,
    const float* __restrict__ t1w, const float* __restrict__ t1b,
    const float* __restrict__ t2w, const float* __restrict__ t2b,
    float* __restrict__ logits) {
  const int b = blockIdx.x, tid = threadIdx.x;
  __shared__ float sp[CD], sh1[CD], sh2[CD / 2], st1[CD / 4];
  __shared__ float red[256];
  __shared__ float sMean, sRstd, sTemp, sll[CE];

  for (int i = tid; i < CD; i += 256) sp[i] = pooled[b * CD + i];
  __syncthreads();

  for (int o = tid; o < CD; o += 256) {
    float s = g1b[o];
    for (int i = 0; i < CD; ++i) s += sp[i] * g1w[i * CD + o];
    sh1[o] = geluf(s);
  }
  __syncthreads();

  // LN over sh1
  float s = 0.0f, ss = 0.0f;
  for (int i = tid; i < CD; i += 256) { s += sh1[i]; ss += sh1[i] * sh1[i]; }
  red[tid] = s; __syncthreads();
  for (int o = 128; o > 0; o >>= 1) { if (tid < o) red[tid] += red[tid + o]; __syncthreads(); }
  if (tid == 0) sMean = red[0] / CD;
  __syncthreads();
  red[tid] = ss; __syncthreads();
  for (int o = 128; o > 0; o >>= 1) { if (tid < o) red[tid] += red[tid + o]; __syncthreads(); }
  if (tid == 0) {
    const float m = sMean;
    sRstd = rsqrtf(red[0] / CD - m * m + 1e-5f);
  }
  __syncthreads();
  for (int i = tid; i < CD; i += 256)
    sh1[i] = (sh1[i] - sMean) * sRstd * lng[i] + lnb[i];
  __syncthreads();

  for (int o = tid; o < CD / 2; o += 256) {
    float v = g2b[o];
    for (int i = 0; i < CD; ++i) v += sh1[i] * g2w[i * (CD / 2) + o];
    sh2[o] = geluf(v);
  }
  for (int o = tid; o < CD / 4; o += 256) {
    float v = t1b[o];
    for (int i = 0; i < CD; ++i) v += sp[i] * t1w[i * (CD / 4) + o];
    st1[o] = geluf(v);
  }
  __syncthreads();

  if (tid < CE) {
    float v = g3b[tid];
    for (int i = 0; i < CD / 2; ++i) v += sh2[i] * g3w[i * CE + tid];
    sll[tid] = v;
  }
  if (tid == 0) {
    float v = t2b[0];
    for (int i = 0; i < CD / 4; ++i) v += st1[i] * t2w[i];
    float sp1 = (v > 20.0f) ? v : log1pf(expf(v));     // softplus
    sTemp = fminf(fmaxf(sp1, 0.1f), 5.0f);
  }
  __syncthreads();
  if (tid < CE) logits[b * CE + tid] = sll[tid] / sTemp;
}

// ------------------------------ top-k + loss -------------------------------

__global__ __launch_bounds__(64) void gate_finalize_kernel(
    const float* __restrict__ logits, float* __restrict__ gates,
    int* __restrict__ idx, float* __restrict__ lossOut) {
  const int tid = threadIdx.x;
  __shared__ float sprobs[CB][CE];
  if (tid < CB) {
    float l[CE];
    for (int e = 0; e < CE; ++e) l[e] = logits[tid * CE + e];
    int i0 = 0;
    for (int e = 1; e < CE; ++e) if (l[e] > l[i0]) i0 = e;
    int i1 = (i0 == 0) ? 1 : 0;
    for (int e = 0; e < CE; ++e) if (e != i0 && l[e] > l[i1]) i1 = e;
    const float e1 = expf(l[i1] - l[i0]);
    gates[tid * CK + 0] = 1.0f / (1.0f + e1);
    gates[tid * CK + 1] = e1 / (1.0f + e1);
    idx[tid * CK + 0] = i0;
    idx[tid * CK + 1] = i1;
    float denom = 0.0f;
    for (int e = 0; e < CE; ++e) denom += expf(l[e] - l[i0]);
    for (int e = 0; e < CE; ++e) sprobs[tid][e] = expf(l[e] - l[i0]) / denom;
  }
  __syncthreads();
  if (tid == 0) {
    float ep[CE], mean = 0.0f;
    for (int e = 0; e < CE; ++e) {
      float s = 0.0f;
      for (int b = 0; b < CB; ++b) s += sprobs[b][e];
      ep[e] = s / CB;
      mean += ep[e];
    }
    mean /= CE;
    float var = 0.0f, ent = 0.0f;
    for (int e = 0; e < CE; ++e) {
      var += (ep[e] - mean) * (ep[e] - mean);
      ent += ep[e] * logf(ep[e] + 1e-8f);
    }
    var /= (CE - 1);                                    // ddof=1
    lossOut[0] = var * CE + 0.1f * (logf((float)CE) + ent);
  }
}

// ------------------------------ expert combine -----------------------------

__global__ void combine_kernel(
    const float* __restrict__ o, const float* __restrict__ g,
    const float* __restrict__ x,
    const float* __restrict__ scale, const float* __restrict__ biasv,
    const float* __restrict__ rw, const float* __restrict__ gates,
    const int* __restrict__ eidx,
    float* __restrict__ y, bhalf* __restrict__ ybf) {
  const long long i = (long long)blockIdx.x * blockDim.x + threadIdx.x;
  if (i >= CBSD) return;
  const int b = (int)(i / ((long long)CS * CD));
  const long long sd = i % ((long long)CS * CD);
  const int d = (int)(i % CD);
  float acc = 0.0f;
#pragma unroll
  for (int k = 0; k < CK; ++k) {
    const int pair = b * CK + k;
    const int e = eidx[pair];
    const float ov = o[(long long)pair * CS * CD + sd];
    const float gv = g[(long long)pair * CS * CD + sd];
    float val = ov * gv * scale[e * CD + d] + biasv[e * CD + d];
    const float r = rw[e];
    val = r * val + (1.0f - r) * x[i];
    acc += gates[pair] * val;
  }
  y[i] = acc;
  ybf[i] = f2bfu(acc);
}

// ------------------------------ host side ----------------------------------

extern "C" void kernel_launch(void* const* d_in, const int* in_sizes, int n_in,
                              void* d_out, int out_size, void* d_ws,
                              size_t ws_size, hipStream_t stream) {
  (void)in_sizes; (void)n_in; (void)out_size; (void)ws_size;
  int ii = 0;
  const float* x          = (const float*)d_in[ii++];  // 0
  const float* attn_in_w  = (const float*)d_in[ii++];
  const float* attn_in_b  = (const float*)d_in[ii++];
  const float* attn_out_w = (const float*)d_in[ii++];
  const float* attn_out_b = (const float*)d_in[ii++];
  const float* g1_w   = (const float*)d_in[ii++];
  const float* g1_b   = (const float*)d_in[ii++];
  const float* g_ln_g = (const float*)d_in[ii++];
  const float* g_ln_b = (const float*)d_in[ii++];
  const float* g2_w   = (const float*)d_in[ii++];
  const float* g2_b   = (const float*)d_in[ii++];
  const float* g3_w   = (const float*)d_in[ii++];
  const float* g3_b   = (const float*)d_in[ii++];
  const float* t1_w   = (const float*)d_in[ii++];
  const float* t1_b   = (const float*)d_in[ii++];
  const float* t2_w   = (const float*)d_in[ii++];
  const float* t2_b   = (const float*)d_in[ii++];     // 16
  const float* in_w    = (const float*)d_in[ii++];    // 17
  const float* in_b    = (const float*)d_in[ii++];
  const float* in_ln_g = (const float*)d_in[ii++];
  const float* in_ln_b = (const float*)d_in[ii++];
  const float* l0_w    = (const float*)d_in[ii++];
  const float* l0_b    = (const float*)d_in[ii++];
  const float* l0_ln_g = (const float*)d_in[ii++];
  const float* l0_ln_b = (const float*)d_in[ii++];
  const float* l1_w    = (const float*)d_in[ii++];
  const float* l1_b    = (const float*)d_in[ii++];
  const float* l1_ln_g = (const float*)d_in[ii++];
  const float* l1_ln_b = (const float*)d_in[ii++];
  const float* out_w   = (const float*)d_in[ii++];
  const float* out_b   = (const float*)d_in[ii++];
  const float* ig1_w   = (const float*)d_in[ii++];
  const float* ig1_b   = (const float*)d_in[ii++];
  const float* ig2_w   = (const float*)d_in[ii++];
  const float* ig2_b   = (const float*)d_in[ii++];
  const float* e_bias  = (const float*)d_in[ii++];    // 35
  const float* e_scale = (const float*)d_in[ii++];
  const float* e_rw    = (const float*)d_in[ii++];
  const float* fu_w   = (const float*)d_in[ii++];     // 38
  const float* fu_b   = (const float*)d_in[ii++];
  const float* norm_g = (const float*)d_in[ii++];
  const float* norm_b = (const float*)d_in[ii++];     // 41

  float* out = (float*)d_out;                          // [B,S,D] + loss

  // workspace bump allocator
  char* wp = (char*)d_ws;
  auto alloc = [&](size_t bytes) -> void* {
    void* r = (void*)wp;
    wp += (bytes + 255) & ~(size_t)255;
    return r;
  };
  bhalf* x_bf     = (bhalf*)alloc(CBSD * 2);
  bhalf* qkv_bf   = (bhalf*)alloc(CBS * 3 * CD * 2);
  bhalf* vt_bf    = (bhalf*)alloc((size_t)CB * CH * CDH * CS * 2);
  bhalf* oattn_bf = (bhalf*)alloc(CBSD * 2);
  float* ctx_f    = (float*)alloc(CBSD * 4);
  float* pooled_f = (float*)alloc((size_t)CB * CD * 4);
  float* logits_f = (float*)alloc((size_t)CB * CE * 4);
  float* gates_f  = (float*)alloc((size_t)CPAIRS * 4);
  int*   idx_i    = (int*)alloc((size_t)CPAIRS * 4);
  bhalf* w_ain  = (bhalf*)alloc((size_t)CD * 3 * CD * 2);
  bhalf* w_aout = (bhalf*)alloc((size_t)CD * CD * 2);
  bhalf* w_in   = (bhalf*)alloc((size_t)CE * CD * CD * 2);
  bhalf* w_l0   = (bhalf*)alloc((size_t)CE * CD * CDFF * 2);
  bhalf* w_l1   = (bhalf*)alloc((size_t)CE * CDFF * CDFF * 2);
  bhalf* w_out  = (bhalf*)alloc((size_t)CE * CDFF * CD * 2);
  bhalf* w_ig1  = (bhalf*)alloc((size_t)CE * CD * (CD / 4) * 2);
  bhalf* w_ig2  = (bhalf*)alloc((size_t)CE * (CD / 4) * CD * 2);
  bhalf* w_fu   = (bhalf*)alloc((size_t)CD * CD * 2);
  float* t_f   = (float*)alloc((size_t)CPAIRS * CS * CD * 4);   // reused as o_f
  bhalf* t_bf  = (bhalf*)alloc((size_t)CPAIRS * CS * CD * 2);
  float* h_f   = (float*)alloc((size_t)CPAIRS * CS * CDFF * 4);
  bhalf* h_bf  = (bhalf*)alloc((size_t)CPAIRS * CS * CDFF * 2);
  bhalf* g1_bf = (bhalf*)alloc((size_t)CPAIRS * CS * (CD / 4) * 2);
  float* g_f   = (float*)alloc((size_t)CPAIRS * CS * CD * 4);
  float* y_f   = (float*)alloc(CBSD * 4);
  bhalf* y_bf  = (bhalf*)alloc(CBSD * 2);
  float* f_f   = (float*)alloc(CBSD * 4);
  float* o_f = t_f;   // alias: t_f dead after input-transform LN

  // --- activation conversion + weight fragment packing (f32 -> bf16) ---
  cvt_kernel<<<1024, 256, 0, stream>>>(x, x_bf, CBSD);
  auto pack = [&](const float* src, bhalf* dst, int Kd, int N, int nE) {
    pack_w_kernel<<<2048, 256, 0, stream>>>(src, dst, Kd, N,
                                            (long long)nE * Kd * N);
  };
  pack(attn_in_w, w_ain, CD, 3 * CD, 1);
  pack(attn_out_w, w_aout, CD, CD, 1);
  pack(in_w, w_in, CD, CD, CE);
  pack(l0_w, w_l0, CD, CDFF, CE);
  pack(l1_w, w_l1, CDFF, CDFF, CE);
  pack(out_w, w_out, CDFF, CD, CE);
  pack(ig1_w, w_ig1, CD, CD / 4, CE);
  pack(ig2_w, w_ig2, CD / 4, CD, CE);
  pack(fu_w, w_fu, CD, CD, 1);

  const long long SD = (long long)CS * CD;
  const long long SF = (long long)CS * CDFF;

  // --- attention: qkv GEMM -> V transpose -> flash attn -> out proj ---
  gemm_bf16_kernel<<<dim3(3 * CD / 128, CBS / 32, 1), 32, 0, stream>>>(
      x_bf, w_ain, attn_in_b, nullptr, nullptr, nullptr, qkv_bf,
      (int)CBS, 3 * CD, CD, 0, 1, 0, 0, 0, 0);
  vtrans_kernel<<<2048, 256, 0, stream>>>(qkv_bf, vt_bf);
  attn_kernel<<<dim3(CS / 16, CH, CB), 32, 0, stream>>>(qkv_bf, vt_bf,
                                                        oattn_bf);
  gemm_bf16_kernel<<<dim3(CD / 128, CBS / 32, 1), 32, 0, stream>>>(
      oattn_bf, w_aout, attn_out_b, x, nullptr, ctx_f, nullptr,
      (int)CBS, CD, CD, 0, 1, 0, 0, 0, 0);

  // --- gate ---
  pooled_kernel<<<(CB * CD) / 256, 256, 0, stream>>>(ctx_f, pooled_f);
  gate_mlp_kernel<<<CB, 256, 0, stream>>>(
      pooled_f, g1_w, g1_b, g_ln_g, g_ln_b, g2_w, g2_b, g3_w, g3_b,
      t1_w, t1_b, t2_w, t2_b, logits_f);
  gate_finalize_kernel<<<1, 64, 0, stream>>>(logits_f, gates_f, idx_i,
                                             out + CBSD);

  // --- experts (16 (b,k) pairs, per-expert weights via idx) ---
  // input transform: Linear -> LN -> GELU
  gemm_bf16_kernel<<<dim3(CD / 128, CS / 32, CPAIRS), 32, 0, stream>>>(
      x_bf, w_in, in_b, nullptr, idx_i, t_f, nullptr,
      CS, CD, CD, SD, CK, (long long)CD * CD, CD, SD, 0);
  ln_kernel<<<CPAIRS * CS, 256, 0, stream>>>(
      t_f, nullptr, in_ln_g, in_ln_b, idx_i, CS, CD, CD, 1, nullptr, t_bf);
  // l0: Linear -> GELU -> LN
  gemm_bf16_kernel<<<dim3(CDFF / 128, CS / 32, CPAIRS), 32, 0, stream>>>(
      t_bf, w_l0, l0_b, nullptr, idx_i, h_f, nullptr,
      CS, CDFF, CD, SD, 1, (long long)CD * CDFF, CDFF, SF, 1);
  ln_kernel<<<CPAIRS * CS, 256, 0, stream>>>(
      h_f, nullptr, l0_ln_g, l0_ln_b, idx_i, CS, CDFF, CDFF, 0, nullptr, h_bf);
  // l1: Linear -> GELU -> LN
  gemm_bf16_kernel<<<dim3(CDFF / 128, CS / 32, CPAIRS), 32, 0, stream>>>(
      h_bf, w_l1, l1_b, nullptr, idx_i, h_f, nullptr,
      CS, CDFF, CDFF, SF, 1, (long long)CDFF * CDFF, CDFF, SF, 1);
  ln_kernel<<<CPAIRS * CS, 256, 0, stream>>>(
      h_f, nullptr, l1_ln_g, l1_ln_b, idx_i, CS, CDFF, CDFF, 0, nullptr, h_bf);
  // out projection
  gemm_bf16_kernel<<<dim3(CD / 128, CS / 32, CPAIRS), 32, 0, stream>>>(
      h_bf, w_out, out_b, nullptr, idx_i, o_f, nullptr,
      CS, CD, CDFF, SF, 1, (long long)CDFF * CD, CD, SD, 0);
  // internal gate: gelu(x@ig1) -> sigmoid(.@ig2)
  gemm_bf16_kernel<<<dim3((CD / 4) / 128, CS / 32, CPAIRS), 32, 0, stream>>>(
      x_bf, w_ig1, ig1_b, nullptr, idx_i, nullptr, g1_bf,
      CS, CD / 4, CD, SD, CK, (long long)CD * (CD / 4), CD / 4,
      (long long)CS * (CD / 4), 1);
  gemm_bf16_kernel<<<dim3(CD / 128, CS / 32, CPAIRS), 32, 0, stream>>>(
      g1_bf, w_ig2, ig2_b, nullptr, idx_i, g_f, nullptr,
      CS, CD, CD / 4, (long long)CS * (CD / 4), 1,
      (long long)(CD / 4) * CD, CD, SD, 2);
  // weighted combine across top-k
  combine_kernel<<<(unsigned)(CBSD / 256), 256, 0, stream>>>(
      o_f, g_f, x, e_scale, e_bias, e_rw, gates_f, idx_i, y_f, y_bf);

  // --- fusion: gelu(y@fu_w+fu_b), then LN(res + x) ---
  gemm_bf16_kernel<<<dim3(CD / 128, CBS / 32, 1), 32, 0, stream>>>(
      y_bf, w_fu, fu_b, nullptr, nullptr, f_f, nullptr,
      (int)CBS, CD, CD, 0, 1, 0, 0, 0, 1);
  ln_kernel<<<(unsigned)CBS, 256, 0, stream>>>(
      f_f, x, norm_g, norm_b, nullptr, 1, 0, CD, 0, out, nullptr);
}
